// RTRBM_970662609534
// MI455X (gfx1250) — compile-verified, matrix operand order checked by reading, not run on hardware
//
#include <hip/hip_runtime.h>
#include <hip/hip_bf16.h>
#include <math.h>

typedef __attribute__((ext_vector_type(16))) __bf16 v16bf;
typedef __attribute__((ext_vector_type(8)))  __bf16 v8bf;
typedef __attribute__((ext_vector_type(8)))  float  v8f;

#define NVIS 784
#define NHID 256
#define TLEN 128
#define BATCH 512
#define KVP  800   // K of W@v padded to a multiple of 32 (784 -> 800)
#define TB   (TLEN * BATCH)
#define VCHUNKS (KVP / 8)       // 100 8-element chunks per column
#define VVALID  (NVIS / 8)      // 98 chunks hold real data, rest zero

// ---------------------------------------------------------------------------
// A/B bf16 WMMA fragment for one lane = two contiguous 16B chunks:
// elements 0..7 at [base + k0 + half*8], elements 8..15 at +16 elements.
// ---------------------------------------------------------------------------
__device__ __forceinline__ v16bf load_frag16(const __bf16* __restrict__ p) {
  v8bf lo = *(const v8bf*)(p);
  v8bf hi = *(const v8bf*)(p + 16);
  return __builtin_shufflevector(lo, hi, 0, 1, 2, 3, 4, 5, 6, 7,
                                 8, 9, 10, 11, 12, 13, 14, 15);
}

// Convert column-slice v_t (fp32, strided) into transposed bf16 vT[BATCH,KVP],
// zero-filling the K pad. Grid-stride over BATCH*VCHUNKS items; one b128
// store per chunk.
__device__ __forceinline__ void convert_v_t(const float* __restrict__ Vin,
                                            int t, int tid, int nthreads,
                                            __bf16* __restrict__ vT) {
  for (int w = tid; w < BATCH * VCHUNKS; w += nthreads) {
    const int n  = w / VCHUNKS;
    const int c  = w - n * VCHUNKS;
    const int k0 = c * 8;
    v8bf pk = {};
    if (c < VVALID) {
#pragma unroll
      for (int e = 0; e < 8; ++e) {
        pk[e] = (__bf16)Vin[(size_t)(k0 + e) * TB + (size_t)t * BATCH + n];
      }
    }
    *(v8bf*)(vT + n * KVP + k0) = pk;
  }
}

// ---------------------------------------------------------------------------
// One-time prep: bf16 W (K-padded), W^T, U; zero r_lag^T; convert v_0.
// ---------------------------------------------------------------------------
__global__ void rtrbm_init(const float* __restrict__ W, const float* __restrict__ U,
                           const float* __restrict__ Vin,
                           __bf16* __restrict__ Wbf, __bf16* __restrict__ Wtbf,
                           __bf16* __restrict__ Ubf, __bf16* __restrict__ rT0,
                           __bf16* __restrict__ vT) {
  const int stride = gridDim.x * blockDim.x;
  const int tid = blockIdx.x * blockDim.x + threadIdx.x;

  for (int i = tid; i < NHID * KVP; i += stride) {           // W -> [NHID x KVP]
    int m = i / KVP, k = i - m * KVP;
    Wbf[i] = (k < NVIS) ? (__bf16)W[m * NVIS + k] : (__bf16)0.0f;
  }
  for (int i = tid; i < NVIS * NHID; i += stride) {          // W^T -> [NVIS x NHID]
    int m = i / NHID, k = i - m * NHID;
    Wtbf[i] = (__bf16)W[k * NVIS + m];
  }
  for (int i = tid; i < NHID * NHID; i += stride) {          // U -> [NHID x NHID]
    Ubf[i] = (__bf16)U[i];
  }
  for (int i = tid; i < BATCH * NHID; i += stride) {         // r_lag^T(t=0) = 0
    rT0[i] = (__bf16)0.0f;
  }
  convert_v_t(Vin, 0, tid, stride, vT);                      // vT <- v_0
}

// ---------------------------------------------------------------------------
// Phase 1: h_mean = sigmoid(W@v_t + U@r_lag + bias); h_sample = noise < h.
// 1x2 tile blocking per wave: 16 M-tiles x 16 N-groups = 256 waves, 32 blocks.
// All operands bf16, all fragments 2x global_load_b128.
// ---------------------------------------------------------------------------
__global__ void __launch_bounds__(256)
rtrbm_hstep(const __bf16* __restrict__ vT,       // [BATCH, KVP]
            const __bf16* __restrict__ Wbf,      // [NHID, KVP]
            const __bf16* __restrict__ Ubf,      // [NHID, NHID]
            const float* __restrict__ bias,      // [NHID]
            const float* __restrict__ noise_h_t, // [NHID, BATCH]
            const __bf16* __restrict__ rT_in,    // [BATCH, NHID]
            __bf16* __restrict__ rT_out,         // [BATCH, NHID]
            __bf16* __restrict__ hsT) {          // [BATCH, NHID]
  const int lane = threadIdx.x & 31;
  const int wave = threadIdx.x >> 5;
  const int tile = blockIdx.x * 8 + wave;        // 0..255
  const int tm   = tile >> 4;                    // 0..15
  const int tg   = tile & 15;                    // 0..15 (pair of 16-col tiles)
  const int half = lane >> 4;
  const int l    = lane & 15;
  const int mrow = tm * 16 + l;
  const int n0   = tg * 32 + l;
  const int n1   = n0 + 16;

  const __bf16* arowW = Wbf + mrow * KVP + half * 8;
  const __bf16* arowU = Ubf + mrow * NHID + half * 8;
  const __bf16* vcol0 = vT + n0 * KVP + half * 8;
  const __bf16* vcol1 = vT + n1 * KVP + half * 8;
  const __bf16* rcol0 = rT_in + n0 * NHID + half * 8;
  const __bf16* rcol1 = rT_in + n1 * NHID + half * 8;

  v8f acc0 = {}, acc1 = {};

  // ---- W @ v_t : K = 800 (25 uniform iterations, zero-padded tail) ----
#pragma unroll 5
  for (int k0 = 0; k0 < KVP; k0 += 32) {
    v16bf a  = load_frag16(arowW + k0);
    v16bf b0 = load_frag16(vcol0 + k0);
    v16bf b1 = load_frag16(vcol1 + k0);
    acc0 = __builtin_amdgcn_wmma_f32_16x16x32_bf16(false, a, false, b0,
                                                   (short)0, acc0, false, false);
    acc1 = __builtin_amdgcn_wmma_f32_16x16x32_bf16(false, a, false, b1,
                                                   (short)0, acc1, false, false);
  }

  // ---- U @ r_lag : K = 256, fully unrolled ----
#pragma unroll
  for (int k0 = 0; k0 < NHID; k0 += 32) {
    v16bf a  = load_frag16(arowU + k0);
    v16bf b0 = load_frag16(rcol0 + k0);
    v16bf b1 = load_frag16(rcol1 + k0);
    acc0 = __builtin_amdgcn_wmma_f32_16x16x32_bf16(false, a, false, b0,
                                                   (short)0, acc0, false, false);
    acc1 = __builtin_amdgcn_wmma_f32_16x16x32_bf16(false, a, false, b1,
                                                   (short)0, acc1, false, false);
  }

  // ---- epilogue: bias + sigmoid + sample; b128 stores into ^T layouts ----
  const int rowbase = tm * 16 + half * 8;        // 8 contiguous rows per lane
#pragma unroll
  for (int tile2 = 0; tile2 < 2; ++tile2) {
    const int n = tile2 ? n1 : n0;
    const v8f& acc = tile2 ? acc1 : acc0;
    v8bf hmv, hsv;
#pragma unroll
    for (int r = 0; r < 8; ++r) {
      const int row = rowbase + r;
      float x  = acc[r] + bias[row];
      float hm = 1.0f / (1.0f + __expf(-x));
      float hs = (noise_h_t[row * BATCH + n] < hm) ? 1.0f : 0.0f;
      hmv[r] = (__bf16)hm;
      hsv[r] = (__bf16)hs;
    }
    *(v8bf*)(rT_out + n * NHID + rowbase) = hmv;
    *(v8bf*)(hsT   + n * NHID + rowbase) = hsv;
  }
}

// ---------------------------------------------------------------------------
// Phase 2: v_mean = sigmoid(W^T@h_sample + b_v); v_sample = noise < v.
// 1x2 blocking: 49 M-tiles x 16 N-groups = 784 waves, 98 blocks.
// Fused tail: convert v_{t+1} into vT for the next h-step.
// ---------------------------------------------------------------------------
__global__ void __launch_bounds__(256)
rtrbm_vstep(const __bf16* __restrict__ hsT,      // [BATCH, NHID]
            const __bf16* __restrict__ Wtbf,     // [NVIS, NHID]
            const float* __restrict__ b_v,       // [NVIS]
            const float* __restrict__ noise_v_t, // [NVIS, BATCH]
            float* __restrict__ out,             // [NVIS, TLEN, BATCH]
            const float* __restrict__ Vin,       // [NVIS, TLEN, BATCH]
            __bf16* __restrict__ vT,             // [BATCH, KVP]
            int t) {
  const int lane = threadIdx.x & 31;
  const int wave = threadIdx.x >> 5;
  const int tile = blockIdx.x * 8 + wave;        // 0..783
  const int tm   = tile >> 4;                    // 0..48
  const int tg   = tile & 15;                    // 0..15
  const int half = lane >> 4;
  const int l    = lane & 15;
  const int mrow = tm * 16 + l;
  const int n0   = tg * 32 + l;
  const int n1   = n0 + 16;

  const __bf16* arow  = Wtbf + mrow * NHID + half * 8;
  const __bf16* hcol0 = hsT + n0 * NHID + half * 8;
  const __bf16* hcol1 = hsT + n1 * NHID + half * 8;

  v8f acc0 = {}, acc1 = {};

  // ---- W^T @ h_sample : K = 256, fully unrolled ----
#pragma unroll
  for (int k0 = 0; k0 < NHID; k0 += 32) {
    v16bf a  = load_frag16(arow + k0);
    v16bf b0 = load_frag16(hcol0 + k0);
    v16bf b1 = load_frag16(hcol1 + k0);
    acc0 = __builtin_amdgcn_wmma_f32_16x16x32_bf16(false, a, false, b0,
                                                   (short)0, acc0, false, false);
    acc1 = __builtin_amdgcn_wmma_f32_16x16x32_bf16(false, a, false, b1,
                                                   (short)0, acc1, false, false);
  }

  // ---- epilogue: bias + sigmoid + sample -> fp32 output [nv, T, B] ----
  const int rowbase = tm * 16 + half * 8;
#pragma unroll
  for (int tile2 = 0; tile2 < 2; ++tile2) {
    const int n = tile2 ? n1 : n0;
    const v8f& acc = tile2 ? acc1 : acc0;
#pragma unroll
    for (int r = 0; r < 8; ++r) {
      const int row = rowbase + r;
      float x  = acc[r] + b_v[row];
      float vm = 1.0f / (1.0f + __expf(-x));
      float vs = (noise_v_t[row * BATCH + n] < vm) ? 1.0f : 0.0f;
      out[(size_t)row * TB + (size_t)t * BATCH + n] = vs;
    }
  }

  // ---- fused: stage bf16 v_{t+1} for the next h-step ----
  if (t + 1 < TLEN) {
    const int tid = blockIdx.x * blockDim.x + threadIdx.x;
    convert_v_t(Vin, t + 1, tid, gridDim.x * blockDim.x, vT);
  }
}

// ---------------------------------------------------------------------------
extern "C" void kernel_launch(void* const* d_in, const int* in_sizes, int n_in,
                              void* d_out, int out_size, void* d_ws, size_t ws_size,
                              hipStream_t stream) {
  const float* Vin     = (const float*)d_in[0];
  const float* W       = (const float*)d_in[1];
  const float* U       = (const float*)d_in[2];
  const float* b_h     = (const float*)d_in[3];
  const float* b_v     = (const float*)d_in[4];
  const float* b_init  = (const float*)d_in[5];
  const float* noise_h = (const float*)d_in[6];
  const float* noise_v = (const float*)d_in[7];
  float* out = (float*)d_out;

  char* ws = (char*)d_ws;
  size_t off = 0;
  auto carve = [&](size_t bytes) -> char* {
    char* p = ws + off;
    off = (off + bytes + 255) & ~(size_t)255;
    return p;
  };
  __bf16* Wbf  = (__bf16*)carve((size_t)NHID * KVP * 2);
  __bf16* Wtbf = (__bf16*)carve((size_t)NVIS * NHID * 2);
  __bf16* Ubf  = (__bf16*)carve((size_t)NHID * NHID * 2);
  __bf16* rT0  = (__bf16*)carve((size_t)BATCH * NHID * 2);
  __bf16* rT1  = (__bf16*)carve((size_t)BATCH * NHID * 2);
  __bf16* hsT  = (__bf16*)carve((size_t)BATCH * NHID * 2);
  __bf16* vT   = (__bf16*)carve((size_t)BATCH * KVP * 2);
  (void)ws_size; (void)in_sizes; (void)n_in; (void)out_size;

  rtrbm_init<<<512, 256, 0, stream>>>(W, U, Vin, Wbf, Wtbf, Ubf, rT0, vT);

  for (int t = 0; t < TLEN; ++t) {
    const float* bias = (t == 0) ? b_init : b_h;
    __bf16* rin  = (t & 1) ? rT1 : rT0;
    __bf16* rout = (t & 1) ? rT0 : rT1;
    rtrbm_hstep<<<32, 256, 0, stream>>>(
        vT, Wbf, Ubf, bias,
        noise_h + (size_t)t * NHID * BATCH,
        rin, rout, hsT);
    rtrbm_vstep<<<98, 256, 0, stream>>>(
        hsT, Wtbf, b_v,
        noise_v + (size_t)t * NVIS * BATCH,
        out, Vin, vT, t);
  }
}
